// MyModel_61933428415326
// MI455X (gfx1250) — matrix operand order repaired and checked by reading, not compiled, not run on hardware
//
#include <hip/hip_runtime.h>
#include <cstdint>
#include <cstddef>

#ifndef __has_builtin
#define __has_builtin(x) 0
#endif

// ---------------- problem dims (from reference) ----------------
constexpr int Bn = 2, Hn = 16, Sn = 2048, Dn = 64;
constexpr int BQ = 64;        // query rows per block (4 waves x 16)
constexpr int WQ = 16;        // query rows per wave
constexpr int KT = 64;        // kv tile (keys per iteration) -> 18 WMMAs/iter
constexpr int NH = KT / 16;   // 16-wide key sub-tiles per iteration
constexpr int NTHREADS = 128; // 4 wave32 waves
constexpr int DP = Dn + 16;   // padded bf16 row (160B: 32B-aligned, 8-bank spread)
constexpr int KP = KT + 16;   // padded bf16 row for V^T / P tiles
#define NEG_INF (-1.0e30f)
#define SM_SCALE 0.125f       // 1/sqrt(64)

typedef __bf16 bf16_t;
typedef __bf16 bf16x16 __attribute__((ext_vector_type(16)));
typedef __bf16 bf16x8  __attribute__((ext_vector_type(8)));
typedef __bf16 bf16x4  __attribute__((ext_vector_type(4)));
typedef float  f32x8   __attribute__((ext_vector_type(8)));
typedef unsigned int u32x4 __attribute__((ext_vector_type(4)));
typedef int i32x4 __attribute__((ext_vector_type(4)));
typedef int i32x8 __attribute__((ext_vector_type(8)));

// Require the CDNA5 Tensor Data Mover path on the DEVICE pass only: a
// successful compile proves tensor_load_to_lds is in the gfx1250 binary.
#if defined(__HIP_DEVICE_COMPILE__)
#if !__has_builtin(__builtin_amdgcn_tensor_load_to_lds) || \
    !__has_builtin(__builtin_amdgcn_s_wait_tensorcnt)
#error "gfx1250 TDM builtins not available on this toolchain (device pass)"
#endif
#endif

// Flat LDS pointer -> LDS byte offset: per ISA aperture rules, the low 32 bits
// of a flat shared address are the wave-relative LDS address.
__device__ __forceinline__ unsigned lds_off(const void* p) {
  return (unsigned)(uintptr_t)p;
}

// Tensor Data Mover 2-D tile load: KT x 64 fp32 tile, row-major, stride 64.
// D# layout per CDNA5 ISA.
__device__ __forceinline__ void tdm_tile_load(const void* gsrc, unsigned ldsbyte,
                                              unsigned rows_avail) {
#if defined(__HIP_DEVICE_COMPILE__)
  unsigned long long ga = (unsigned long long)gsrc;
  u32x4 g0;
  g0[0] = 1u;                                               // count=1 (valid), user mode
  g0[1] = ldsbyte;                                          // lds_addr
  g0[2] = (unsigned)ga;                                     // global_addr[31:0]
  g0[3] = ((unsigned)(ga >> 32) & 0x1FFFFFFu) | (2u << 30); // addr[56:32] | type=2
  i32x8 g1;
  g1[0] = (int)(2u << 16);                                  // data_size=4B
  g1[1] = (int)((unsigned)Dn << 16);                        // tensor_dim0 lo16
  g1[2] = (int)((rows_avail & 0xFFFFu) << 16);              // dim0 hi=0 | tensor_dim1 lo
  g1[3] = (int)(((rows_avail >> 16) & 0xFFFFu) | ((unsigned)Dn << 16)); // dim1 hi | tile_dim0
  g1[4] = (int)KT;                                          // tile_dim1=KT, tile_dim2=0
  g1[5] = (int)Dn;                                          // tensor_dim0_stride=64
  g1[6] = 0;
  g1[7] = 0;
  i32x4 z4 = {0, 0, 0, 0};
#if __clang_major__ >= 23
  i32x8 z8 = {0, 0, 0, 0, 0, 0, 0, 0};
  __builtin_amdgcn_tensor_load_to_lds(g0, g1, z4, z4, z8, 0);
#else
  __builtin_amdgcn_tensor_load_to_lds(g0, g1, z4, z4, 0);
#endif
#else
  (void)gsrc; (void)ldsbyte; (void)rows_avail;
#endif
}

__device__ __forceinline__ void tdm_wait0() {
#if defined(__HIP_DEVICE_COMPILE__)
  __builtin_amdgcn_s_wait_tensorcnt(0);
#endif
}

__global__ __launch_bounds__(NTHREADS, 1)
void doc_flash_attn_wmma(const float* __restrict__ qg, const float* __restrict__ kg,
                         const float* __restrict__ vg, const int* __restrict__ docg,
                         float* __restrict__ outg) {
  // LDS: double-buffered raw fp32 K/V tiles (TDM targets) + padded bf16 tiles.
  __shared__ __align__(16) float kraw[2][KT][Dn];          // 32 KB
  __shared__ __align__(16) float vraw[2][KT][Dn];          // 32 KB
  __shared__ __align__(32) bf16_t kbuf[KT][DP];            // 10 KB, [key][d]
  __shared__ __align__(32) bf16_t vtb[Dn][KP];             // 10 KB, [d][key]
  __shared__ __align__(32) bf16_t sb[NTHREADS / 32][WQ][KP]; // 10 KB, P staging
  __shared__ int dts[KT];                                  // tile doc ids

  const int tid  = threadIdx.x;
  const int wid  = tid >> 5;
  const int lane = tid & 31;
  const int lh   = lane & 15;       // lane % 16 (N column / M row index)
  const int hi16 = (lane >> 4) & 1; // half-wave select

  const int nqt = Sn / BQ;
  const int qt  = blockIdx.x % nqt;
  const int bh  = blockIdx.x / nqt;
  const size_t base = (size_t)bh * Sn * Dn;
  const float* Qb = qg + base;
  const float* Kb = kg + base;
  const float* Vb = vg + base;
  float* Ob = outg + base;

  const int q0blk = qt * BQ;
  const int q0 = q0blk + wid * WQ;

  // ---- Q tile -> bf16 A-layout registers (two K=32 chunks over D=64) ----
  bf16x16 qa0, qa1;
  {
    const float* qrow = Qb + (size_t)(q0 + lh) * Dn;
    const int kb0 = hi16 ? 16 : 0;
#pragma unroll
    for (int e = 0; e < 8; ++e) {
      qa0[e]     = (bf16_t)qrow[kb0 + e];
      qa0[8 + e] = (bf16_t)qrow[kb0 + 8 + e];
      qa1[e]     = (bf16_t)qrow[32 + kb0 + e];
      qa1[8 + e] = (bf16_t)qrow[32 + kb0 + 8 + e];
    }
  }

  // All-ones B matrix: row-sum of P via the matrix pipe (P x 1 -> every C
  // column holds the row sums, in the same lane/VGPR layout as srow).
  bf16x16 onesb;
#pragma unroll
  for (int e = 0; e < 16; ++e) onesb[e] = (bf16_t)1.0f;

  // Per-lane doc ids for this lane's 8 C-matrix rows + wave-uniform doc info.
  int qdoc[8];
#pragma unroll
  for (int r = 0; r < 8; ++r) qdoc[r] = docg[q0 + hi16 * 8 + r];
  const int wdoc0 = docg[q0];
  const bool wuni = (wdoc0 == docg[q0 + WQ - 1]); // wave's 16 queries same doc?

  // ---- contiguous valid KV range (doc_ids sorted => block-diagonal mask) ----
  const int dq_lo = docg[q0blk];
  const int dq_hi = docg[q0blk + BQ - 1];
  int lo = 0, hi = Sn;
  while (lo < hi) { int mid = (lo + hi) >> 1; if (docg[mid] < dq_lo) lo = mid + 1; else hi = mid; }
  const int kv_lo = lo & ~(KT - 1);
  int lo2 = 0, hi2 = Sn;
  while (lo2 < hi2) { int mid = (lo2 + hi2) >> 1; if (docg[mid] <= dq_hi) lo2 = mid + 1; else hi2 = mid; }
  const int ntiles = (lo2 - kv_lo + KT - 1) / KT;

  // ---- online softmax state ----
  const f32x8 zero8 = {0, 0, 0, 0, 0, 0, 0, 0};
  float mrow[8], srow[8];
  f32x8 o[4] = {zero8, zero8, zero8, zero8};
#pragma unroll
  for (int r = 0; r < 8; ++r) { mrow[r] = NEG_INF; srow[r] = 0.f; }

  auto stage = [&](int t, int bufi) {
    const int kv0 = kv_lo + t * KT;
    if (wid == 0) {
      tdm_tile_load(Kb + (size_t)kv0 * Dn, lds_off(&kraw[bufi][0][0]), (unsigned)(Sn - kv0));
      tdm_tile_load(Vb + (size_t)kv0 * Dn, lds_off(&vraw[bufi][0][0]), (unsigned)(Sn - kv0));
    }
  };

  if (ntiles > 0) {
    stage(0, 0);
    for (int t = 0; t < ntiles; ++t) {
      const int cur = t & 1;
      const int kv0 = kv_lo + t * KT;
      if (wid == 0) tdm_wait0();
      __syncthreads();                       // raw[cur] landed; prev compute done
      if (t + 1 < ntiles) stage(t + 1, cur ^ 1);

      // ---- vectorized fp32 -> bf16 convert (K row-major; V transposed) ----
      {
        const float4* kr4 = (const float4*)&kraw[cur][0][0];
        for (int i = tid; i < KT * Dn / 4; i += NTHREADS) {  // 8 iters
          const float4 x = kr4[i];
          bf16x4 y;
          y[0] = (bf16_t)x.x; y[1] = (bf16_t)x.y; y[2] = (bf16_t)x.z; y[3] = (bf16_t)x.w;
          *(bf16x4*)&kbuf[i >> 4][(i & 15) * 4] = y;
        }
        for (int i = tid; i < Dn * (KT / 4); i += NTHREADS) { // 8 iters
          const int d = i & 63;        // consecutive lanes -> consecutive d
          const int kg = (i >> 6) * 4; // 4 keys per store
          bf16x4 y;
          y[0] = (bf16_t)vraw[cur][kg + 0][d];
          y[1] = (bf16_t)vraw[cur][kg + 1][d];
          y[2] = (bf16_t)vraw[cur][kg + 2][d];
          y[3] = (bf16_t)vraw[cur][kg + 3][d];
          *(bf16x4*)&vtb[d][kg] = y;
        }
        if (tid < KT) dts[tid] = docg[kv0 + tid];
      }
      __syncthreads();

      // ---- S = Q K^T : four 16x16 score tiles, K-dim 64 in two x32 WMMAs ----
      const int db = hi16 ? 16 : 0;
      f32x8 s[NH];
#pragma unroll
      for (int h = 0; h < NH; ++h) {
        const bf16_t* kr = &kbuf[16 * h + lh][0];
        bf16x16 blo = *(const bf16x16*)(kr + db);
        bf16x16 bhi = *(const bf16x16*)(kr + 32 + db);
        s[h] = zero8;
        s[h] = __builtin_amdgcn_wmma_f32_16x16x32_bf16(false, qa0, false, blo, (short)0, s[h], false, false);
        s[h] = __builtin_amdgcn_wmma_f32_16x16x32_bf16(false, qa1, false, bhi, (short)0, s[h], false, false);
      }

      // ---- doc mask (uniform fast path) + online softmax ----
      float p[NH][8], tmx[8];
#pragma unroll
      for (int r = 0; r < 8; ++r) tmx[r] = -__builtin_inff();
#pragma unroll
      for (int h = 0; h < NH; ++h) {
        const int th0 = dts[16 * h];
        const bool nomask = wuni && (th0 == dts[16 * h + 15]) && (th0 == wdoc0);
        if (nomask) {
#pragma unroll
          for (int r = 0; r < 8; ++r) {
            const float a = s[h][r] * SM_SCALE;
            p[h][r] = a; tmx[r] = fmaxf(tmx[r], a);
          }
        } else {
          const int kdh = dts[16 * h + lh];
#pragma unroll
          for (int r = 0; r < 8; ++r) {
            const float a = (qdoc[r] == kdh) ? s[h][r] * SM_SCALE : -__builtin_inff();
            p[h][r] = a; tmx[r] = fmaxf(tmx[r], a);
          }
        }
      }
      // row-max across the 16-lane half-groups (xor 1,2,4,8 stays in-group)
#pragma unroll
      for (int off = 1; off < 16; off <<= 1) {
#pragma unroll
        for (int r = 0; r < 8; ++r) tmx[r] = fmaxf(tmx[r], __shfl_xor(tmx[r], off, 32));
      }
#pragma unroll
      for (int r = 0; r < 8; ++r) {
        // clamp keeps nm finite even for fully-masked rows, so exp needs no
        // predicate: masked p = -inf  ->  exp(-inf - nm) = 0.
        const float nm = fmaxf(fmaxf(mrow[r], tmx[r]), NEG_INF);
        const float corr = __expf(mrow[r] - nm);
        mrow[r] = nm;
#pragma unroll
        for (int h = 0; h < NH; ++h) p[h][r] = __expf(p[h][r] - nm);
        srow[r] *= corr;
#pragma unroll
        for (int nc = 0; nc < 4; ++nc) o[nc][r] *= corr;
      }

      // ---- P: C-layout -> A-layout via per-wave LDS scratch ----
      {
        const int mb = hi16 * 8;
#pragma unroll
        for (int h = 0; h < NH; ++h) {
#pragma unroll
          for (int r = 0; r < 8; ++r) sb[wid][mb + r][16 * h + lh] = (bf16_t)p[h][r];
        }
      }
      asm volatile("s_wait_dscnt 0" ::: "memory"); // same-wave LDS RAW ordering

      // ---- O += P V, and srow += P x 1 on the matrix pipe ----
      {
        const int kb0 = hi16 ? 16 : 0;
        const bf16_t* prow = &sb[wid][lh][0];
        bf16x8 a0 = *(const bf16x8*)(prow + kb0);
        bf16x8 a1 = *(const bf16x8*)(prow + kb0 + 8);
        bf16x8 a2 = *(const bf16x8*)(prow + 32 + kb0);
        bf16x8 a3 = *(const bf16x8*)(prow + 32 + kb0 + 8);
        bf16x16 paA, paB;
#pragma unroll
        for (int e = 0; e < 8; ++e) {
          paA[e] = a0[e]; paA[8 + e] = a1[e];
          paB[e] = a2[e]; paB[8 + e] = a3[e];
        }
        // softmax denominator: every C column of P x ones = row sum
        f32x8 rsv = zero8;
        rsv = __builtin_amdgcn_wmma_f32_16x16x32_bf16(false, paA, false, onesb, (short)0, rsv, false, false);
        rsv = __builtin_amdgcn_wmma_f32_16x16x32_bf16(false, paB, false, onesb, (short)0, rsv, false, false);
#pragma unroll
        for (int nc = 0; nc < 4; ++nc) {
          const bf16_t* vrow = &vtb[16 * nc + lh][0];
          bf16x16 v0 = *(const bf16x16*)(vrow + db);
          bf16x16 v1 = *(const bf16x16*)(vrow + 32 + db);
          o[nc] = __builtin_amdgcn_wmma_f32_16x16x32_bf16(false, paA, false, v0, (short)0, o[nc], false, false);
          o[nc] = __builtin_amdgcn_wmma_f32_16x16x32_bf16(false, paB, false, v1, (short)0, o[nc], false, false);
        }
#pragma unroll
        for (int r = 0; r < 8; ++r) srow[r] += rsv[r];
      }
      __syncthreads();                        // compute done before next convert
    }
  }

  // ---- normalize + store O ----
#pragma unroll
  for (int r = 0; r < 8; ++r) {
    const float inv = (srow[r] > 0.f) ? (1.f / srow[r]) : 0.f;
    const int m = hi16 * 8 + r;
    float* orow = Ob + (size_t)(q0 + m) * Dn;
    orow[lh]      = o[0][r] * inv;
    orow[16 + lh] = o[1][r] * inv;
    orow[32 + lh] = o[2][r] * inv;
    orow[48 + lh] = o[3][r] * inv;
  }
}

extern "C" void kernel_launch(void* const* d_in, const int* in_sizes, int n_in,
                              void* d_out, int out_size, void* d_ws, size_t ws_size,
                              hipStream_t stream) {
  (void)in_sizes; (void)n_in; (void)out_size; (void)d_ws; (void)ws_size;
  const float* q   = (const float*)d_in[0];
  const float* k   = (const float*)d_in[1];
  const float* v   = (const float*)d_in[2];
  const int*   doc = (const int*)d_in[3];
  float* out = (float*)d_out;
  dim3 grid((Sn / BQ) * Bn * Hn);   // 1024 blocks
  dim3 block(NTHREADS);             // 4 wave32 waves
  doc_flash_attn_wmma<<<grid, block, 0, stream>>>(q, k, v, doc, out);
}
